// SDM_7284264534537
// MI455X (gfx1250) — compile-verified, matrix operand order checked by reading, not run on hardware
//
#include <hip/hip_runtime.h>

typedef __attribute__((ext_vector_type(16))) _Float16     v16h;
typedef __attribute__((ext_vector_type(8)))  float        v8f;
typedef __attribute__((ext_vector_type(8)))  unsigned int v8u;

#define Bn 4
#define Hh 192
#define Ww 192
#define HW (Hh*Ww)

union H2 { _Float16 h[2]; unsigned int u; };

template <int ACT>
__device__ __forceinline__ float actf(float v) {
  if constexpr (ACT == 1) return v > 0.f ? v : 0.f;        // relu
  if constexpr (ACT == 2) return v > 0.f ? v : 0.1f * v;   // leaky relu 0.1
  return v;
}

// ---------------------------------------------------------------------------
// f32 -> f16 weight pack ((Cout,Cin,3,3)/(Cout,Cin,1,1) flatten to the exact
// [Cout][K] order the GEMM wants, so pack == convert).
// ---------------------------------------------------------------------------
__global__ void cvt_f16(const float* __restrict__ s, _Float16* __restrict__ d, int n) {
  int i = blockIdx.x * 256 + threadIdx.x;
  if (i < n) d[i] = (_Float16)s[i];
}

// ---------------------------------------------------------------------------
// Implicit-GEMM 3x3 conv, pad=1, via v_wmma_f32_16x16x32_f16.
// Block = 128 thr (4 waves). Tile: 16 pixels (fixed n,h; w0..w0+15) x 64 couts.
// grid = (W/16, H, B * Cout/64).  K = Cin*9 (576 or 1152, both %32==0).
// LDS holds the tile pre-im2col'd and pre-paired: sAu[k/2*16+m] = {k, k+1}
// packed f16 -> A fragment build = 8x ds_load_b32 with shift-add addressing.
// ---------------------------------------------------------------------------
template <int ACT, bool SKIP>
__global__ void __launch_bounds__(128)
conv3x3_wmma(const float* __restrict__ in, const _Float16* __restrict__ wpk,
             const float* __restrict__ bias, const float* __restrict__ skip,
             float* __restrict__ out, int Cin, int Cout)
{
  __shared__ unsigned int sAu[9216];                // up to (1152/2)*16 dwords (36KB)
  const int w0 = blockIdx.x * 16;
  const int h  = blockIdx.y;
  const int groups   = Cout >> 6;
  const int n_img    = (int)blockIdx.z / groups;
  const int coutBase = ((int)blockIdx.z % groups) * 64;
  const int tid = threadIdx.x;
  const int K  = Cin * 9;

  // stage: im2col into paired-dword layout
  const int dwTotal = (K >> 1) * 16;
  for (int d = tid; d < dwTotal; d += 128) {
    int kp = d >> 4, m = d & 15;
    H2 x;
#pragma unroll
    for (int e = 0; e < 2; ++e) {
      int k = (kp << 1) + e;
      int c = k / 9, t = k - c * 9;
      int di = t / 3, dj = t - di * 3;
      int gh = h + di - 1, gw = w0 + m + dj - 1;
      float v = 0.f;
      if (gh >= 0 && gh < Hh && gw >= 0 && gw < Ww)
        v = in[(((size_t)n_img * Cin + c) * Hh + gh) * Ww + gw];
      x.h[e] = (_Float16)actf<ACT>(v);
    }
    sAu[d] = x.u;
  }
  __syncthreads();

  const int wave = tid >> 5, lane = tid & 31;
  const int ln = lane & 15, hb = lane >> 4;         // ln = pixel row (A) / cout col (B)
  const int n0 = coutBase + wave * 16;
  const unsigned int* wrowu = (const unsigned int*)(wpk + (size_t)(n0 + ln) * K);

  v8f acc = {};
  const int KD = K >> 1;
#pragma unroll 2
  for (int kd0 = 0; kd0 < KD; kd0 += 16) {
    v8u au, bu;
#pragma unroll
    for (int v = 0; v < 8; ++v) {
      int dw = kd0 + ((v >> 2) << 3) + (hb << 2) + (v & 3);
      au[v] = sAu[dw * 16 + ln];
      bu[v] = wrowu[dw];
    }
    acc = __builtin_amdgcn_wmma_f32_16x16x32_f16(
        false, __builtin_bit_cast(v16h, au), false, __builtin_bit_cast(v16h, bu),
        (short)0, acc, false, false);
  }

  const int cout = n0 + ln;
  const float bv = bias[cout];
#pragma unroll
  for (int v = 0; v < 8; ++v) {
    int px = v + (hb << 3);
    size_t oi = (((size_t)n_img * Cout + cout) * Hh + h) * Ww + (w0 + px);
    float r = acc[v] + bv;
    if constexpr (SKIP) r += skip[oi];
    out[oi] = r;
  }
}

// ---------------------------------------------------------------------------
// 1x1 conv (Cout=64) as GEMM: M = B*HW pixels, K = Cin. grid = (B*HW/16).
// ---------------------------------------------------------------------------
__global__ void __launch_bounds__(128)
conv1x1_wmma(const float* __restrict__ in, const _Float16* __restrict__ wpk,
             const float* __restrict__ bias, float* __restrict__ out,
             int Cin, int Cout)
{
  __shared__ unsigned int sAu[1024];                // (128/2)*16 dwords
  const int g0    = blockIdx.x * 16;                // HW%16==0 -> no batch crossing
  const int n_img = g0 / HW;
  const int p0    = g0 - n_img * HW;
  const int tid = threadIdx.x;

  const int dwTotal = (Cin >> 1) * 16;
  for (int d = tid; d < dwTotal; d += 128) {
    int kp = d >> 4, m = d & 15;
    H2 x;
    x.h[0] = (_Float16)in[((size_t)n_img * Cin + (kp << 1)) * HW + p0 + m];
    x.h[1] = (_Float16)in[((size_t)n_img * Cin + (kp << 1) + 1) * HW + p0 + m];
    sAu[d] = x.u;
  }
  __syncthreads();

  const int wave = tid >> 5, lane = tid & 31;
  const int ln = lane & 15, hb = lane >> 4;
  const int n0 = wave * 16;                         // 4 waves x 16 = 64 couts
  const unsigned int* wrowu = (const unsigned int*)(wpk + (size_t)(n0 + ln) * Cin);

  v8f acc = {};
  const int KD = Cin >> 1;
#pragma unroll 2
  for (int kd0 = 0; kd0 < KD; kd0 += 16) {
    v8u au, bu;
#pragma unroll
    for (int v = 0; v < 8; ++v) {
      int dw = kd0 + ((v >> 2) << 3) + (hb << 2) + (v & 3);
      au[v] = sAu[dw * 16 + ln];
      bu[v] = wrowu[dw];
    }
    acc = __builtin_amdgcn_wmma_f32_16x16x32_f16(
        false, __builtin_bit_cast(v16h, au), false, __builtin_bit_cast(v16h, bu),
        (short)0, acc, false, false);
  }

  const int cout = n0 + ln;
  const float bv = bias[cout];
#pragma unroll
  for (int v = 0; v < 8; ++v) {
    int px = v + (hb << 3);
    out[((size_t)n_img * Cout + cout) * HW + p0 + px] = acc[v] + bv;
  }
}

// ---------------------------------------------------------------------------
// Fused: filt = kg_conv(lrelu(x_kg)) (1x1, 64->576) kept in LDS, then
// out_rgb = sum_t unfold3x3(rgb)*filt.  Avoids the 340MB filter tensor.
// grid = (W/16, H, B), block = 128.
// ---------------------------------------------------------------------------
__global__ void __launch_bounds__(128)
kgconv_dyn_fused(const float* __restrict__ xkg, const _Float16* __restrict__ wpk,
                 const float* __restrict__ bias, const float* __restrict__ rgb,
                 float* __restrict__ outR)
{
  __shared__ unsigned int sXu[512];                 // (64/2)*16 dwords
  __shared__ float        sF[576 * 16];             // filter tile: [q][m]
  const int w0 = blockIdx.x * 16, h = blockIdx.y, n = blockIdx.z;
  const int tid = threadIdx.x;

  for (int d = tid; d < 512; d += 128) {
    int kp = d >> 4, m = d & 15;
    H2 x;
#pragma unroll
    for (int e = 0; e < 2; ++e) {
      float v = xkg[(((size_t)n * 64 + (kp << 1) + e) * Hh + h) * Ww + w0 + m];
      x.h[e] = (_Float16)(v > 0.f ? v : 0.1f * v);
    }
    sXu[d] = x.u;
  }
  __syncthreads();

  const int wave = tid >> 5, lane = tid & 31;
  const int ln = lane & 15, hb = lane >> 4;

  for (int it = 0; it < 9; ++it) {                  // 36 n-tiles / 4 waves
    int n0 = (wave + 4 * it) * 16;
    const unsigned int* wrowu = (const unsigned int*)(wpk + (size_t)(n0 + ln) * 64);
    v8f acc = {};
#pragma unroll
    for (int kd0 = 0; kd0 < 32; kd0 += 16) {
      v8u au, bu;
#pragma unroll
      for (int v = 0; v < 8; ++v) {
        int dw = kd0 + ((v >> 2) << 3) + (hb << 2) + (v & 3);
        au[v] = sXu[dw * 16 + ln];
        bu[v] = wrowu[dw];
      }
      acc = __builtin_amdgcn_wmma_f32_16x16x32_f16(
          false, __builtin_bit_cast(v16h, au), false, __builtin_bit_cast(v16h, bu),
          (short)0, acc, false, false);
    }
    int q = n0 + ln;
    float bv = bias[q];
#pragma unroll
    for (int v = 0; v < 8; ++v) sF[q * 16 + (v + (hb << 3))] = acc[v] + bv;
  }
  __syncthreads();

  for (int i = tid; i < 64 * 16; i += 128) {        // dynamic local conv
    int c = i >> 4, m = i & 15;
    float s = 0.f;
#pragma unroll
    for (int t = 0; t < 9; ++t) {
      int gh = h + t / 3 - 1, gw = w0 + m + t % 3 - 1;
      float v = 0.f;
      if (gh >= 0 && gh < Hh && gw >= 0 && gw < Ww)
        v = rgb[(((size_t)n * 64 + c) * Hh + gh) * Ww + gw];
      s += v * sF[(c * 9 + t) * 16 + m];
    }
    outR[(((size_t)n * 64 + c) * Hh + h) * Ww + w0 + m] = s;
  }
}

// ---------------------------------------------------------------------------
// Glue kernels
// ---------------------------------------------------------------------------
__global__ void bilinear_down(const float* __restrict__ in, float* __restrict__ dn) {
  size_t i = (size_t)blockIdx.x * 256 + threadIdx.x;        // Bn*64*48*48
  int bc = (int)(i / 2304), rem = (int)(i - (size_t)bc * 2304);
  int oy = rem / 48, ox = rem - oy * 48;
  const float* s = in + (size_t)bc * HW;
  int y0 = 4 * oy + 1, x0 = 4 * ox + 1;                     // src=(o+.5)*4-.5 -> w=.5
  dn[i] = 0.25f * (s[y0 * Ww + x0] + s[y0 * Ww + x0 + 1] +
                   s[(y0 + 1) * Ww + x0] + s[(y0 + 1) * Ww + x0 + 1]);
}

__global__ void diff_up(const float* __restrict__ dep, const float* __restrict__ dn,
                        float* __restrict__ diff) {
  size_t i = (size_t)blockIdx.x * 256 + threadIdx.x;        // Bn*64*HW
  int bc = (int)(i / HW), p = (int)(i - (size_t)bc * HW);
  int y = p / Ww, x = p - y * Ww;
  float sy = y * 0.25f - 0.375f, sx = x * 0.25f - 0.375f;
  float fy = floorf(sy), fx = floorf(sx);
  int y0 = (int)fy, x0 = (int)fx;
  float wy = sy - fy, wx = sx - fx;
  int y0c = min(max(y0, 0), 47), y1c = min(max(y0 + 1, 0), 47);
  int x0c = min(max(x0, 0), 47), x1c = min(max(x0 + 1, 0), 47);
  const float* s = dn + (size_t)bc * 2304;
  float v00 = s[y0c * 48 + x0c], v01 = s[y0c * 48 + x1c];
  float v10 = s[y1c * 48 + x0c], v11 = s[y1c * 48 + x1c];
  float up = (1.f - wy) * ((1.f - wx) * v00 + wx * v01) +
             wy * ((1.f - wx) * v10 + wx * v11);
  diff[i] = fabsf(dep[i] - up);
}

__global__ void gap_kernel(const float* __restrict__ in, float* __restrict__ y) {
  __shared__ float red[256];
  int b = blockIdx.x;
  const float* p = in + (size_t)b * HW;
  float acc = 0.f;
  for (int i = threadIdx.x; i < HW; i += 256) acc += p[i];
  red[threadIdx.x] = acc;
  __syncthreads();
  for (int s = 128; s > 0; s >>= 1) {
    if (threadIdx.x < s) red[threadIdx.x] += red[threadIdx.x + s];
    __syncthreads();
  }
  if (threadIdx.x == 0) y[b] = red[0] * (1.f / HW);
}

__global__ void ca_kernel(const float* __restrict__ y, const float* __restrict__ wd,
                          const float* __restrict__ bd, const float* __restrict__ wu,
                          const float* __restrict__ bu, float* __restrict__ s,
                          int C, int RD) {
  __shared__ float z[16];
  int n = blockIdx.x, c = threadIdx.x;
  if (c < RD) {
    float acc = bd[c];
    for (int i = 0; i < C; ++i) acc += wd[c * C + i] * y[n * C + i];
    z[c] = acc > 0.f ? acc : 0.f;
  }
  __syncthreads();
  float acc = bu[c];
  for (int r = 0; r < RD; ++r) acc += wu[c * RD + r] * z[r];
  s[n * C + c] = 1.f / (1.f + __expf(-acc));
}

__global__ void scale_skip(const float* __restrict__ r, const float* __restrict__ s,
                           const float* __restrict__ x, float* __restrict__ out) {
  size_t i = (size_t)blockIdx.x * 256 + threadIdx.x;
  int nc = (int)(i / HW);
  out[i] = r[i] * s[nc] + x[i];
}

__global__ void avgmax_kernel(const float* __restrict__ diff, float* __restrict__ amap) {
  size_t i = (size_t)blockIdx.x * 256 + threadIdx.x;        // Bn*HW
  int n = (int)(i / HW), p = (int)(i - (size_t)n * HW);
  const float* s = diff + (size_t)n * 64 * HW + p;
  float a = 0.f, m = -3.4e38f;
  for (int c = 0; c < 64; ++c) { float v = s[(size_t)c * HW]; a += v; m = fmaxf(m, v); }
  amap[((size_t)n * 2) * HW + p]     = a * (1.f / 64.f);
  amap[((size_t)n * 2 + 1) * HW + p] = m;
}

__global__ void conv_du_att(const float* __restrict__ amap, const float* __restrict__ w,
                            const float* __restrict__ b, float* __restrict__ att) {
  size_t i = (size_t)blockIdx.x * 256 + threadIdx.x;        // Bn*64*HW
  int n = (int)(i / ((size_t)64 * HW));
  size_t rem = i - (size_t)n * 64 * HW;
  int c = (int)(rem / HW), p = (int)(rem - (size_t)c * HW);
  int y = p / Ww, x = p - y * Ww;
  float acc = b[c];
  for (int ch = 0; ch < 2; ++ch)
#pragma unroll
    for (int t = 0; t < 9; ++t) {
      int gh = y + t / 3 - 1, gw = x + t % 3 - 1;
      if (gh >= 0 && gh < Hh && gw >= 0 && gw < Ww)
        acc += amap[((size_t)(n * 2 + ch)) * HW + gh * Ww + gw] * w[(c * 2 + ch) * 9 + t];
    }
  att[i] = 1.f / (1.f + __expf(-acc));
}

__global__ void rgb2_kernel(const float* __restrict__ att, const float* __restrict__ rgb,
                            const float* __restrict__ beta, float* __restrict__ R) {
  size_t i = (size_t)blockIdx.x * 256 + threadIdx.x;
  R[i] = att[i] * rgb[i] + beta[0] * R[i];
}

__global__ void cat_kernel(const float* __restrict__ dep, const float* __restrict__ rgb2,
                           float* __restrict__ D) {
  size_t i = (size_t)blockIdx.x * 256 + threadIdx.x;        // Bn*128*HW
  int n = (int)(i / ((size_t)128 * HW));
  size_t rem = i - (size_t)n * 128 * HW;
  int c = (int)(rem / HW), p = (int)(rem - (size_t)c * HW);
  D[i] = (c < 64) ? dep[((size_t)n * 64 + c) * HW + p]
                  : rgb2[((size_t)n * 64 + (c - 64)) * HW + p];
}

// ---------------------------------------------------------------------------
extern "C" void kernel_launch(void* const* d_in, const int* in_sizes, int n_in,
                              void* d_out, int out_size, void* d_ws, size_t ws_size,
                              hipStream_t stream)
{
  (void)in_sizes; (void)n_in; (void)out_size; (void)ws_size;
  const float* dep   = (const float*)d_in[0];
  const float* rgb   = (const float*)d_in[1];
  const float* du_w  = (const float*)d_in[4];
  const float* du_b  = (const float*)d_in[5];
  const float* beta  = (const float*)d_in[30];

  const size_t P = 9437184ull;                       // 4*64*192*192
  float* out_re = (float*)d_out;
  float* att    = out_re + P;
  float* diff   = out_re + 2 * P;                    // diff_intra output slot

  // Workspace (phase-aliased, ~230MB):
  float* ws = (float*)d_ws;
  float* bufA = ws;                 // P   : diff_f (KG rcab input x)
  float* bufB = ws + P;             // P   : kg r1 / t
  float* bufC = ws + 2 * P;         // P   : kg r2 / x_kg
  float* bufDown = ws + 3 * P;      // 589824: down; later avg/max map (294912)
  float* bufR = ws + 3 * P + 655360;// P   : out_rgb -> rgb_feat2 (in place)
  float* bufD = ws;                 // 2P  : cat  (aliases A,B; KG dead)
  float* bufE = ws + 2 * P;         // 2P  : rg r1/t (aliases C,down,R-low; all dead)
  float* bufF = ws + 4 * P + 655360;// 2P  : rg r2 / out_rg
  float* bufY = ws + 6 * P + 655360;// 512 : GAP result
  float* bufS = bufY + 512;         // 512 : CA sigmoid scales
  _Float16* wh = (_Float16*)(bufS + 512);
  _Float16* w_diff = wh;                 // 64*576
  _Float16* w_kg1  = w_diff + 36864;
  _Float16* w_kg2  = w_kg1 + 36864;
  _Float16* w_kgt  = w_kg2 + 36864;
  _Float16* w_kgc  = w_kgt + 36864;      // 576*64
  _Float16* w_rg1  = w_kgc + 36864;      // 128*1152
  _Float16* w_rg2  = w_rg1 + 147456;
  _Float16* w_rgt  = w_rg2 + 147456;
  _Float16* w_re   = w_rgt + 147456;     // 64*128

  auto cvt = [&](int idx, _Float16* dst, int n) {
    cvt_f16<<<(n + 255) / 256, 256, 0, stream>>>((const float*)d_in[idx], dst, n);
  };
  cvt(2, w_diff, 36864);  cvt(6, w_kg1, 36864);   cvt(8, w_kg2, 36864);
  cvt(14, w_kgt, 36864);  cvt(16, w_kgc, 36864);  cvt(18, w_rg1, 147456);
  cvt(20, w_rg2, 147456); cvt(26, w_rgt, 147456); cvt(28, w_re, 8192);

  // down-up bilinear residual -> diff_intra (output slot)
  bilinear_down<<<2304, 256, 0, stream>>>(dep, bufDown);
  diff_up<<<36864, 256, 0, stream>>>(dep, bufDown, diff);

  dim3 g64(12, 192, 4), g128(12, 192, 8), blk(128);

  // diff_intra_f = conv_diff(lrelu(diff))
  conv3x3_wmma<2, false><<<g64, blk, 0, stream>>>(diff, w_diff, (const float*)d_in[3],
                                                  nullptr, bufA, 64, 64);
  // KGenerator residual group (C=64)
  conv3x3_wmma<0, false><<<g64, blk, 0, stream>>>(bufA, w_kg1, (const float*)d_in[7],
                                                  nullptr, bufB, 64, 64);
  conv3x3_wmma<1, false><<<g64, blk, 0, stream>>>(bufB, w_kg2, (const float*)d_in[9],
                                                  nullptr, bufC, 64, 64);   // r2
  gap_kernel<<<256, 256, 0, stream>>>(bufC, bufY);
  ca_kernel<<<4, 64, 0, stream>>>(bufY, (const float*)d_in[10], (const float*)d_in[11],
                                  (const float*)d_in[12], (const float*)d_in[13],
                                  bufS, 64, 4);
  scale_skip<<<36864, 256, 0, stream>>>(bufC, bufS, bufA, bufB);            // rcab out
  conv3x3_wmma<0, true><<<g64, blk, 0, stream>>>(bufB, w_kgt, (const float*)d_in[15],
                                                 bufA, bufC, 64, 64);       // x_kg
  // fused kg_conv(1x1, 64->576) + dynamic local conv -> out_rgb
  kgconv_dyn_fused<<<dim3(12, 192, 4), blk, 0, stream>>>(
      bufC, w_kgc, (const float*)d_in[17], rgb, bufR);

  // spatial attention + fuse
  avgmax_kernel<<<576, 256, 0, stream>>>(diff, bufDown);
  conv_du_att<<<36864, 256, 0, stream>>>(bufDown, du_w, du_b, att);
  rgb2_kernel<<<36864, 256, 0, stream>>>(att, rgb, beta, bufR);             // rgb_feat2
  cat_kernel<<<73728, 256, 0, stream>>>(dep, bufR, bufD);                   // (b,128,h,w)

  // Rgb residual group (C2=128)
  conv3x3_wmma<0, false><<<g128, blk, 0, stream>>>(bufD, w_rg1, (const float*)d_in[19],
                                                   nullptr, bufE, 128, 128);
  conv3x3_wmma<1, false><<<g128, blk, 0, stream>>>(bufE, w_rg2, (const float*)d_in[21],
                                                   nullptr, bufF, 128, 128); // r2
  gap_kernel<<<512, 256, 0, stream>>>(bufF, bufY);
  ca_kernel<<<4, 128, 0, stream>>>(bufY, (const float*)d_in[22], (const float*)d_in[23],
                                   (const float*)d_in[24], (const float*)d_in[25],
                                   bufS, 128, 8);
  scale_skip<<<73728, 256, 0, stream>>>(bufF, bufS, bufD, bufE);            // rcab out
  conv3x3_wmma<0, true><<<g128, blk, 0, stream>>>(bufE, w_rgt, (const float*)d_in[27],
                                                  bufD, bufF, 128, 128);    // out_rg
  // out_re = conv_re(out_rg) (1x1, 128->64)
  conv1x1_wmma<<<9216, blk, 0, stream>>>(bufF, w_re, (const float*)d_in[29],
                                         out_re, 128, 64);
}
